// MultiScaleRetentionCompat_16441134809484
// MI455X (gfx1250) — matrix operand order repaired and checked by reading, not compiled, not run on hardware
//
#include <hip/hip_runtime.h>
#include <hip/hip_bf16.h>
#include <math.h>

typedef unsigned short u16;
typedef __attribute__((ext_vector_type(16))) __bf16 v16bf;
typedef __attribute__((ext_vector_type(8)))  __bf16 v8bf;
typedef __attribute__((ext_vector_type(8)))  float  v8f;
typedef __attribute__((ext_vector_type(4)))  unsigned int v4u;
typedef __attribute__((ext_vector_type(8)))  int v8i;
typedef __attribute__((ext_vector_type(4)))  int v4i;

#define Bn   4
#define Sn   4096
#define Dn   1024
#define Hn   16
#define HDn  64
#define Mn   (Bn * Sn)          // 16384 rows
#define Kn   Dn                 // 1024 reduction dim
#define NCH  16                 // scan chunks
#define CLEN 256                // chunk length (NCH*CLEN == Sn)
#define LN_EPS 1e-5f

#if __has_builtin(__builtin_amdgcn_tensor_load_to_lds) && __has_builtin(__builtin_amdgcn_s_wait_tensorcnt)
#define USE_TDM 1
#else
#define USE_TDM 0
#endif

// ---------------- fp32 -> bf16 (RNE) ----------------
__device__ __forceinline__ u16 f2bf(float f) {
  union { float f; unsigned u; } c; c.f = f;
  unsigned u = c.u;
  u += 0x7fffu + ((u >> 16) & 1u);
  return (u16)(u >> 16);
}

__global__ __launch_bounds__(256) void cvt_x_kernel(const float* __restrict__ x,
                                                    u16* __restrict__ xb) {
  int i = blockIdx.x * blockDim.x + threadIdx.x;       // over Mn*Dn/4
  float4 v = ((const float4*)x)[i];
  ushort4 o;
  o.x = f2bf(v.x); o.y = f2bf(v.y); o.z = f2bf(v.z); o.w = f2bf(v.w);
  ((ushort4*)xb)[i] = o;
}

// Wcat rows: [0,1024)=Wq, [1024,2048)=Wv, [2048,3072)=Wg, [3072,4096)=Wo
__global__ __launch_bounds__(256) void cvt_w_kernel(const float* __restrict__ Wq,
                                                    const float* __restrict__ Wv,
                                                    const float* __restrict__ Wg,
                                                    const float* __restrict__ Wo,
                                                    u16* __restrict__ Wcat) {
  int i = blockIdx.x * blockDim.x + threadIdx.x;       // over 4*Dn*Dn/4
  int which = i >> 18;                                  // Dn*Dn/4 = 2^18
  int w4    = i & ((1 << 18) - 1);
  const float* src = (which == 0) ? Wq : (which == 1) ? Wv : (which == 2) ? Wg : Wo;
  float4 v = ((const float4*)src)[w4];
  ushort4 o;
  o.x = f2bf(v.x); o.y = f2bf(v.y); o.z = f2bf(v.z); o.w = f2bf(v.w);
  ((ushort4*)Wcat)[i] = o;
}

// ---------------- bf16 WMMA GEMM: C[m,n] = sum_k A[m,k] * W[n,k] ----------------
// Block tile 128x128, BK=32, 8 waves (4 in M x 2 in N), wave tile 32x64.
// LDS tiles staged by the Tensor Data Mover (double buffered); TDM pad feature
// produces the bank-conflict-avoiding stride of 40 bf16 (64B row + 16B pad).
#define LDT 40   // padded LDS stride in bf16 elems (32 + 8)

__device__ __forceinline__ unsigned lds_offset(const void* p) {
  // Generic-space LDS addresses carry the wave-relative offset in the low 32 bits
  return (unsigned)(uintptr_t)p;
}

#if USE_TDM
// 2D tile load: 128 rows x 32 bf16, row stride Kn elements, into LDS with
// pad_interval = 16 DWORDs (one row), pad_amount = 4 DWORDs (16B) -> LDT 40.
__device__ __forceinline__ void tdm_load_tile(const u16* tile, unsigned tensor_rows,
                                              unsigned lds_byte_off) {
  unsigned long long ga = (unsigned long long)(uintptr_t)tile;
  v4u g0;
  g0.x = 1u;                                                   // count=1 (valid), user mode
  g0.y = lds_byte_off;                                         // lds_addr
  g0.z = (unsigned)ga;                                         // global_addr[31:0]
  g0.w = (unsigned)((ga >> 32) & 0x01ffffffu) | 0x80000000u;   // global_addr[56:32] | type=2
  v8i g1;
  g1[0] = (1 << 16) | (1 << 20) | (3 << 22) | (3 << 25);       // data_size=2B, pad_en, intv=16dw, amt=4dw
  g1[1] = (int)((Kn & 0xffffu) << 16);                         // tensor_dim0 lo16 (abar addr = 0)
  g1[2] = (int)(((unsigned)Kn >> 16) | ((tensor_rows & 0xffffu) << 16)); // dim0 hi16 | dim1 lo16
  g1[3] = (int)((tensor_rows >> 16) | (32u << 16));            // dim1 hi16 | tile_dim0=32
  g1[4] = 128;                                                 // tile_dim1=128, tile_dim2=0
  g1[5] = (int)Kn;                                             // tensor_dim0_stride lo32
  g1[6] = 0;                                                   // stride hi16 | dim1_stride lo16
  g1[7] = 0;
  v4i g2; g2.x = 1; g2.y = 1; g2.z = 0; g2.w = 0;              // tensor_dim2=1, dim3=1
  v4i g3; g3.x = 0; g3.y = 0x00010000; g3.z = 0; g3.w = 0;     // tensor_dim4=1
  v8i g4;                                                      // extra group (this toolchain's
  g4[0] = 0; g4[1] = 0; g4[2] = 0; g4[3] = 0;                  // 6-arg form) — zero filled
  g4[4] = 0; g4[5] = 0; g4[6] = 0; g4[7] = 0;
  __builtin_amdgcn_tensor_load_to_lds(g0, g1, g2, g3, g4, 0);
}
#endif

template <int NCOLS>
__global__ __launch_bounds__(256) void gemm_bf16_kernel(const u16* __restrict__ A,
                                                        const u16* __restrict__ W,
                                                        float* __restrict__ C) {
  __shared__ __align__(16) u16 As[2][128 * LDT];
  __shared__ __align__(16) u16 Bs[2][128 * LDT];

  const int bm   = blockIdx.y * 128;
  const int bn   = blockIdx.x * 128;
  const int t    = threadIdx.x;
  const int lane = t & 31;
  const int wave = t >> 5;
  const int wm   = (wave & 3) * 32;   // wave M offset in tile
  const int wn   = (wave >> 2) * 64;  // wave N offset in tile

  v8f acc[2][4];
  v8f zero = {0.f, 0.f, 0.f, 0.f, 0.f, 0.f, 0.f, 0.f};
#pragma unroll
  for (int i = 0; i < 2; ++i)
#pragma unroll
    for (int j = 0; j < 4; ++j) acc[i][j] = zero;

  // Fragment addressing per CDNA5 ISA layouts
  const int ar = lane & 15;           // A row within 16
  const int ak = (lane >> 4) * 8;     // A K base: lanes<16 -> {0..7,16..23}, >=16 -> {8..15,24..31}
  const int bc = lane & 15;           // B col within 16
  const int bk = (lane >> 4) * 8;     // B K base: lanes<16 -> {0..7,16..23}, >=16 -> {8..15,24..31}

#if USE_TDM
  const unsigned ldsA[2] = { lds_offset(&As[0][0]), lds_offset(&As[1][0]) };
  const unsigned ldsB[2] = { lds_offset(&Bs[0][0]), lds_offset(&Bs[1][0]) };
  if (wave == 0) {
    tdm_load_tile(A + (size_t)bm * Kn, (unsigned)Mn, ldsA[0]);
    tdm_load_tile(W + (size_t)bn * Kn, (unsigned)NCOLS, ldsB[0]);
  }
#endif

  const int nkt = Kn / 32;
  for (int kt = 0; kt < nkt; ++kt) {
    const int cur = kt & 1;
#if USE_TDM
    if (wave == 0) {
      if (kt + 1 < nkt) {
        const int k1 = (kt + 1) * 32;
        tdm_load_tile(A + (size_t)bm * Kn + k1, (unsigned)Mn, ldsA[cur ^ 1]);
        tdm_load_tile(W + (size_t)bn * Kn + k1, (unsigned)NCOLS, ldsB[cur ^ 1]);
        __builtin_amdgcn_s_wait_tensorcnt(2);   // stage kt's two TDM ops are done
      } else {
        __builtin_amdgcn_s_wait_tensorcnt(0);
      }
    }
    __syncthreads();   // release consumers; only wave0's signal gates on TDM data
#else
    const int k0 = kt * 32;
#pragma unroll
    for (int it = 0; it < 2; ++it) {
      int idx = t + it * 256;
      int row = idx >> 2;
      int kc  = (idx & 3) * 8;
      *(v8bf*)(&As[cur][0] + row * LDT + kc) =
          *(const v8bf*)(A + (size_t)(bm + row) * Kn + k0 + kc);
      *(v8bf*)(&Bs[cur][0] + row * LDT + kc) =
          *(const v8bf*)(W + (size_t)(bn + row) * Kn + k0 + kc);
    }
    __syncthreads();
#endif

    const u16* Ab = &As[cur][0];
    const u16* Bb = &Bs[cur][0];
    v16bf af[2], bfrag[4];
#pragma unroll
    for (int mf = 0; mf < 2; ++mf) {
      const u16* p = Ab + (wm + mf * 16 + ar) * LDT + ak;
      v8bf lo = *(const v8bf*)p;            // VGPR0-3
      v8bf hi = *(const v8bf*)(p + 16);     // VGPR4-7 (K +16)
      af[mf] = __builtin_shufflevector(lo, hi, 0,1,2,3,4,5,6,7,8,9,10,11,12,13,14,15);
    }
#pragma unroll
    for (int nf = 0; nf < 4; ++nf) {
      const u16* p = Bb + (wn + nf * 16 + bc) * LDT + bk;
      v8bf lo = *(const v8bf*)p;            // K bk..bk+7
      v8bf hi = *(const v8bf*)(p + 16);     // K bk+16..bk+23
      bfrag[nf] = __builtin_shufflevector(lo, hi, 0,1,2,3,4,5,6,7,8,9,10,11,12,13,14,15);
    }

#pragma unroll
    for (int mf = 0; mf < 2; ++mf)
#pragma unroll
      for (int nf = 0; nf < 4; ++nf)
        acc[mf][nf] = __builtin_amdgcn_wmma_f32_16x16x32_bf16(
            false, af[mf], false, bfrag[nf], (short)0, acc[mf][nf], false, false);
    __syncthreads();   // all reads of 'cur' done before stage kt+2 overwrites it
  }

  // epilogue: f32 C/D layout — lane 0-15: M=r, lane 16-31: M=8+r; N = lane&15
  const int cr0 = (lane >> 4) * 8;
  const int cc  = lane & 15;
#pragma unroll
  for (int mf = 0; mf < 2; ++mf)
#pragma unroll
    for (int nf = 0; nf < 4; ++nf) {
      float* cp = C + (size_t)(bm + wm + mf * 16 + cr0) * NCOLS + bn + wn + nf * 16 + cc;
#pragma unroll
      for (int r = 0; r < 8; ++r) cp[(size_t)r * NCOLS] = acc[mf][nf][r];
    }
}

// ---------------- retention: chunked linear scan state[t] = lam*state[t-1] + v[t] ----
// C1 layout: [m][3072] = q | v | gpre. chunk arrays layout: [(b*NCH + c)*Dn + d]
__global__ __launch_bounds__(256) void ret_pass1(const float* __restrict__ C1,
                                                 const float* __restrict__ decay,
                                                 float* __restrict__ chunkEnd) {
  int gid = blockIdx.x * blockDim.x + threadIdx.x;   // Bn*NCH*Dn
  int d    = gid & (Dn - 1);
  int rest = gid >> 10;
  int c    = rest & (NCH - 1);
  int b    = rest >> 4;
  float lam = decay[d >> 6];
  const float* vp = C1 + (size_t)(b * Sn + c * CLEN) * 3072 + Dn + d;
  float s = 0.f;
  for (int i = 0; i < CLEN; ++i) s = lam * s + vp[(size_t)i * 3072];
  chunkEnd[gid] = s;
}

__global__ __launch_bounds__(256) void ret_pass2(const float* __restrict__ decay,
                                                 const float* __restrict__ chunkEnd,
                                                 float* __restrict__ chunkInit) {
  int gid = blockIdx.x * blockDim.x + threadIdx.x;   // Bn*Dn
  int d = gid & (Dn - 1);
  int b = gid >> 10;
  float lam  = decay[d >> 6];
  float lamL = powf(lam, (float)CLEN);
  float s = 0.f;
  for (int c = 0; c < NCH; ++c) {
    size_t idx = (size_t)(b * NCH + c) * Dn + d;
    chunkInit[idx] = s;
    s = lamL * s + chunkEnd[idx];
  }
}

__global__ __launch_bounds__(256) void ret_pass3(const float* __restrict__ C1,
                                                 const float* __restrict__ decay,
                                                 const float* __restrict__ chunkInit,
                                                 float* __restrict__ y0) {
  int gid = blockIdx.x * blockDim.x + threadIdx.x;   // Bn*NCH*Dn
  int d    = gid & (Dn - 1);
  int rest = gid >> 10;
  int c    = rest & (NCH - 1);
  int b    = rest >> 4;
  float lam = decay[d >> 6];
  float s = chunkInit[gid];
  const float* base = C1 + (size_t)(b * Sn + c * CLEN) * 3072;
  float* yo = y0 + (size_t)(b * Sn + c * CLEN) * Dn + d;
  for (int i = 0; i < CLEN; ++i) {
    const float* rowp = base + (size_t)i * 3072;
    s = lam * s + rowp[Dn + d];          // v
    yo[(size_t)i * Dn] = rowp[d] * s;    // q * state
  }
}

// ---------------- LayerNorm + silu gate, write bf16 activations ----------------
__global__ __launch_bounds__(256) void ln_gate_kernel(const float* __restrict__ y0,
                                                      const float* __restrict__ C1,
                                                      const float* __restrict__ gamma,
                                                      const float* __restrict__ beta,
                                                      u16* __restrict__ yg) {
  __shared__ float s_sum[8], s_sq[8];
  int m = blockIdx.x;
  int t = threadIdx.x;
  float4 v = ((const float4*)(y0 + (size_t)m * Dn))[t];
  float sum = v.x + v.y + v.z + v.w;
  float sq  = v.x * v.x + v.y * v.y + v.z * v.z + v.w * v.w;
#pragma unroll
  for (int off = 16; off > 0; off >>= 1) {
    sum += __shfl_xor(sum, off);
    sq  += __shfl_xor(sq,  off);
  }
  if ((t & 31) == 0) { s_sum[t >> 5] = sum; s_sq[t >> 5] = sq; }
  __syncthreads();
  float tot = 0.f, totsq = 0.f;
#pragma unroll
  for (int i = 0; i < 8; ++i) { tot += s_sum[i]; totsq += s_sq[i]; }
  float mean = tot * (1.0f / Dn);
  float var  = totsq * (1.0f / Dn) - mean * mean;
  float rstd = rsqrtf(var + LN_EPS);

  const float* gp = C1 + (size_t)m * 3072 + 2 * Dn + t * 4;
  ushort4 o;
  float e[4] = {v.x, v.y, v.z, v.w};
  u16 ob[4];
#pragma unroll
  for (int j = 0; j < 4; ++j) {
    int d = t * 4 + j;
    float yn = (e[j] - mean) * rstd * gamma[d] + beta[d];
    float g  = gp[j];
    float sg = g / (1.0f + __expf(-g));   // silu
    ob[j] = f2bf(yn * sg);
  }
  o.x = ob[0]; o.y = ob[1]; o.z = ob[2]; o.w = ob[3];
  ((ushort4*)(yg + (size_t)m * Dn))[t] = o;
}

// ---------------- launch ----------------
extern "C" void kernel_launch(void* const* d_in, const int* in_sizes, int n_in,
                              void* d_out, int out_size, void* d_ws, size_t ws_size,
                              hipStream_t stream) {
  const float* x     = (const float*)d_in[0];
  const float* decay = (const float*)d_in[1];
  const float* Wq    = (const float*)d_in[2];
  const float* Wv    = (const float*)d_in[3];
  const float* Wo    = (const float*)d_in[4];
  const float* Wg    = (const float*)d_in[5];
  const float* gamma = (const float*)d_in[6];
  const float* beta  = (const float*)d_in[7];
  float* out = (float*)d_out;

  char* p = (char*)d_ws;
  u16*   xb    = (u16*)p;   p += (size_t)Mn * Dn * 2;          // 32 MB
  u16*   Wcat  = (u16*)p;   p += (size_t)4 * Dn * Dn * 2;      // 8 MB
  float* C1    = (float*)p; p += (size_t)Mn * 3 * Dn * 4;      // 192 MB
  float* y0    = (float*)p; p += (size_t)Mn * Dn * 4;          // 64 MB
  u16*   yg    = (u16*)p;   p += (size_t)Mn * Dn * 2;          // 32 MB
  float* cEnd  = (float*)p; p += (size_t)Bn * NCH * Dn * 4;
  float* cInit = (float*)p;

  // 1. precision conversion
  cvt_x_kernel<<<(Mn * Dn / 4) / 256, 256, 0, stream>>>(x, xb);
  cvt_w_kernel<<<(4 * Dn * Dn / 4) / 256, 256, 0, stream>>>(Wq, Wv, Wg, Wo, Wcat);

  // 2. fused projection GEMM: C1 = xb x [Wq|Wv|Wg]^T  (N = 3072)
  gemm_bf16_kernel<3 * Dn><<<dim3(3 * Dn / 128, Mn / 128), 256, 0, stream>>>(xb, Wcat, C1);

  // 3. retention scan (3-pass chunked) fused with q*state
  ret_pass1<<<(Bn * NCH * Dn) / 256, 256, 0, stream>>>(C1, decay, cEnd);
  ret_pass2<<<(Bn * Dn) / 256, 256, 0, stream>>>(decay, cEnd, cInit);
  ret_pass3<<<(Bn * NCH * Dn) / 256, 256, 0, stream>>>(C1, decay, cInit, y0);

  // 4. LayerNorm + silu gate -> bf16
  ln_gate_kernel<<<Mn, 256, 0, stream>>>(y0, C1, gamma, beta, yg);

  // 5. output projection: out = yg x Wo^T
  gemm_bf16_kernel<Dn><<<dim3(Dn / 128, Mn / 128), 256, 0, stream>>>(
      yg, Wcat + (size_t)3 * Dn * Dn, out);
}
